// GATv2Encoder_89146341195956
// MI455X (gfx1250) — matrix operand order repaired and checked by reading, not compiled, not run on hardware
//
#include <hip/hip_runtime.h>
#include <cstring>
#include <cstdint>

// ---------------------------------------------------------------------------
// Problem constants (match reference)
// ---------------------------------------------------------------------------
#define NNODES 10000
#define NEDGE  80000
#define EALL   (NEDGE + NNODES)
#define DIN    512
#define HHEADS 8
#define CDIM   128
#define HDIM   1024
#define OUTD   256
#define SLOPE  0.2f
#define LNEPS  1e-5f

typedef unsigned short ushort_t;

// ---------------------------------------------------------------------------
// WMMA types (CDNA5 / gfx1250, wave32)
// ---------------------------------------------------------------------------
typedef __attribute__((ext_vector_type(16))) __bf16 v16bf;
typedef __attribute__((ext_vector_type(8)))  float  v8f;

union FragAB { v16bf v; uint4 q[2]; };
union FragC  { v8f   v; float f[8]; };

// ---------------------------------------------------------------------------
// Small helpers
// ---------------------------------------------------------------------------
__device__ __forceinline__ ushort_t f2bf(float f) {
    unsigned u = __float_as_uint(f);
    unsigned r = (u + 0x7FFFu + ((u >> 16) & 1u)) >> 16;
    return (ushort_t)r;
}

// order-preserving float <-> signed int encoding (for atomicMax on floats)
__device__ __forceinline__ int fenc(float f) {
    int b = __float_as_int(f);
    return b >= 0 ? b : (int)(0x80000000u - (unsigned)b);
}
__device__ __forceinline__ float fdec(int e) {
    int b = e >= 0 ? e : (int)(0x80000000u - (unsigned)e);
    return __int_as_float(b);
}

__global__ void fill_f32_kernel(float* p, float v, int n) {
    int t = blockIdx.x * blockDim.x + threadIdx.x;
    if (t < n) p[t] = v;
}
__global__ void fill_i32_kernel(int* p, int v, int n) {
    int t = blockIdx.x * blockDim.x + threadIdx.x;
    if (t < n) p[t] = v;
}

// fp32 -> bf16 convert (row-major, same layout)
__global__ void conv_bf16_kernel(const float* __restrict__ x,
                                 ushort_t* __restrict__ o, int n) {
    int t = blockIdx.x * blockDim.x + threadIdx.x;
    if (t < n) o[t] = f2bf(x[t]);
}

// W[K x Nn] fp32 -> Wt[Nn x K] bf16 (transpose + convert)
__global__ void wconv_kernel(const float* __restrict__ W,
                             ushort_t* __restrict__ Wt, int K, int Nn) {
    int t = blockIdx.x * blockDim.x + threadIdx.x;
    if (t >= K * Nn) return;
    int n = t / K;
    int k = t - n * K;
    Wt[t] = f2bf(W[(size_t)k * Nn + n]);
}

// ---------------------------------------------------------------------------
// bf16 WMMA GEMM:  C[M x Nn] = A[M x K](bf16) * Wt[Nn x K](bf16)^T + bias
// act: 0 = none, 1 = ELU.  Block tile 128x64, 8 waves of 32x32, BK=32.
// ---------------------------------------------------------------------------
#define BM 128
#define BN 64
#define BK 32
#define LDT 40   // LDS row stride in ushorts (padded, keeps 16B alignment)

__global__ __launch_bounds__(256)
void gemm_bf16_kernel(const ushort_t* __restrict__ A,
                      const ushort_t* __restrict__ Wt,
                      const float* __restrict__ bias,
                      float* __restrict__ C,
                      int M, int Nn, int K, int act)
{
    __shared__ ushort_t As[BM * LDT];
    __shared__ ushort_t Bs[BN * LDT];

    const int tid  = threadIdx.x;
    const int lane = tid & 31;
    const int wv   = tid >> 5;
    const int wm   = wv & 3;    // 4 wave rows (M)
    const int wn   = wv >> 2;   // 2 wave cols (N)
    const int mBase = blockIdx.x * BM;
    const int nBase = blockIdx.y * BN;

    FragC acc[2][2];
#pragma unroll
    for (int mi = 0; mi < 2; ++mi)
#pragma unroll
        for (int ni = 0; ni < 2; ++ni)
#pragma unroll
            for (int r = 0; r < 8; ++r) acc[mi][ni].f[r] = 0.f;

    const int rsel = lane & 15;           // row (A frag) / col (B frag)
    const int khA  = (lane >> 4) * 8;     // A: K = khA..khA+7 and khA+16..khA+23
    const int kbB  = (lane >> 4) * 8;     // B: K = kbB..kbB+7 and kbB+16..kbB+23

    for (int kk = 0; kk < K; kk += BK) {
        // ---- stage A tile (128 x 32 bf16): 2 threads per row, 16 halves each
        {
            int row = tid >> 1;
            int cg  = (tid & 1) * 16;
            uint4 d0 = {0, 0, 0, 0}, d1 = {0, 0, 0, 0};
            int gm = mBase + row;
            if (gm < M) {
                const uint4* p = (const uint4*)(A + (size_t)gm * K + kk + cg);
                d0 = p[0];
                d1 = p[1];
            }
            *(uint4*)&As[row * LDT + cg]     = d0;
            *(uint4*)&As[row * LDT + cg + 8] = d1;
        }
        // ---- stage B tile (64 x 32 bf16)
        if (tid < 128) {
            int row = tid >> 1;
            int cg  = (tid & 1) * 16;
            const uint4* p = (const uint4*)(Wt + (size_t)(nBase + row) * K + kk + cg);
            *(uint4*)&Bs[row * LDT + cg]     = p[0];
            *(uint4*)&Bs[row * LDT + cg + 8] = p[1];
        }
        __syncthreads();

        FragAB fa[2], fb[2];
#pragma unroll
        for (int mi = 0; mi < 2; ++mi) {
            const ushort_t* base = &As[(wm * 32 + mi * 16 + rsel) * LDT];
            fa[mi].q[0] = *(const uint4*)(base + khA);
            fa[mi].q[1] = *(const uint4*)(base + khA + 16);
        }
#pragma unroll
        for (int ni = 0; ni < 2; ++ni) {
            const ushort_t* base = &Bs[(wn * 32 + ni * 16 + rsel) * LDT];
            fb[ni].q[0] = *(const uint4*)(base + kbB);
            fb[ni].q[1] = *(const uint4*)(base + kbB + 16);
        }
#pragma unroll
        for (int mi = 0; mi < 2; ++mi)
#pragma unroll
            for (int ni = 0; ni < 2; ++ni)
                acc[mi][ni].v = __builtin_amdgcn_wmma_f32_16x16x32_bf16(
                    false, fa[mi].v, false, fb[ni].v,
                    (short)0, acc[mi][ni].v, false, false);
        __syncthreads();
    }

    // ---- epilogue: C/D layout — lane(0..15)=N, +16 lanes -> M+8, VGPR r = M%8
    const int mAdd = (lane >> 4) * 8;
#pragma unroll
    for (int mi = 0; mi < 2; ++mi) {
#pragma unroll
        for (int ni = 0; ni < 2; ++ni) {
            int col = nBase + wn * 32 + ni * 16 + (lane & 15);
            float bb = bias[col];
#pragma unroll
            for (int rr = 0; rr < 8; ++rr) {
                int row = mBase + wm * 32 + mi * 16 + mAdd + rr;
                if (row < M) {
                    float v = acc[mi][ni].f[rr] + bb;
                    if (act) v = v > 0.f ? v : expf(v) - 1.f;
                    C[(size_t)row * Nn + col] = v;
                }
            }
        }
    }
}

// ---------------------------------------------------------------------------
// Edge stage: GATv2 attention
// ---------------------------------------------------------------------------
// one wave per edge; lane covers HD/32 contiguous features; per-head shfl reduce
template <int HD, int C>
__global__ __launch_bounds__(256)
void edge_logits_kernel(const float* __restrict__ xl, const float* __restrict__ xr,
                        const float* __restrict__ att,
                        const int* __restrict__ srcE, const int* __restrict__ dstE,
                        int E0, int Eall,
                        float* __restrict__ ebuf, int* __restrict__ emaxEnc)
{
    constexpr int H   = HD / C;
    constexpr int FPL = HD / 32;   // features per lane
    constexpr int LPH = C / FPL;   // lanes per head
    int wave = (blockIdx.x * blockDim.x + threadIdx.x) >> 5;
    int lane = threadIdx.x & 31;
    if (wave >= Eall) return;
    int s, d;
    if (wave < E0) { s = srcE[wave]; d = dstE[wave]; }
    else           { s = wave - E0;  d = s; }

    int f0   = lane * FPL;
    int head = f0 / C;
    const float* pl = xl  + (size_t)s * HD + f0;
    const float* pr = xr  + (size_t)d * HD + f0;
    const float* pa = att + f0;
    float acc = 0.f;
#pragma unroll
    for (int j = 0; j < FPL; ++j) {
        float v = pl[j] + pr[j];
        v = v > 0.f ? v : SLOPE * v;
        acc += v * pa[j];
    }
#pragma unroll
    for (int w = 1; w < LPH; w <<= 1) acc += __shfl_xor(acc, w, 32);
    if ((lane & (LPH - 1)) == 0) {
        ebuf[(size_t)wave * H + head] = acc;
        atomicMax(&emaxEnc[d * H + head], fenc(acc));
    }
}

// ex = exp(e - max); accumulate softmax denominator
__global__ void edge_exp_kernel(float* __restrict__ ebuf,
                                const int* __restrict__ emaxEnc,
                                float* __restrict__ denom,
                                const int* __restrict__ dstE,
                                int E0, int Eall, int H)
{
    int t = blockIdx.x * blockDim.x + threadIdx.x;
    if (t >= Eall * H) return;
    int e  = t / H;
    int hh = t - e * H;
    int d  = (e < E0) ? dstE[e] : (e - E0);
    float ex = expf(ebuf[t] - fdec(emaxEnc[d * H + hh]));
    ebuf[t] = ex;
    atomicAdd(&denom[d * H + hh], ex);
}

// aggr[dst] += alpha * xl[src]; 4 features per thread
__global__ __launch_bounds__(256)
void edge_scatter_kernel(const float* __restrict__ xl,
                         const float* __restrict__ ebuf,
                         const float* __restrict__ denom,
                         const int* __restrict__ srcE, const int* __restrict__ dstE,
                         int E0, int Eall, int HD, int C,
                         float* __restrict__ aggr)
{
    int t   = blockIdx.x * blockDim.x + threadIdx.x;
    int tpe = HD >> 2;
    int e   = t / tpe;
    if (e >= Eall) return;
    int f = (t - e * tpe) << 2;
    int s, d;
    if (e < E0) { s = srcE[e]; d = dstE[e]; }
    else        { s = e - E0;  d = s; }
    int H  = HD / C;
    int hh = f / C;
    float alpha = ebuf[(size_t)e * H + hh] / denom[d * H + hh];
    float4 v = *(const float4*)(xl + (size_t)s * HD + f);
    float* o = aggr + (size_t)d * HD + f;
    atomicAdd(o + 0, alpha * v.x);
    atomicAdd(o + 1, alpha * v.y);
    atomicAdd(o + 2, alpha * v.z);
    atomicAdd(o + 3, alpha * v.w);
}

// out = LN( elu(aggr + bc) + res ) * g + be ; one block (256 thr) per node row
__global__ __launch_bounds__(256)
void elu_res_ln_kernel(const float* __restrict__ aggr, const float* __restrict__ bc,
                       const float* __restrict__ res,  const float* __restrict__ g,
                       const float* __restrict__ be,   float* __restrict__ out, int D)
{
    int row = blockIdx.x;
    int tid = threadIdx.x;
    int per = D >> 8;   // D / 256 (1 or 4)
    float vals[4];
    float s = 0.f, s2 = 0.f;
    for (int i = 0; i < per; ++i) {
        int f = tid + (i << 8);
        float v = aggr[(size_t)row * D + f] + bc[f];
        v = v > 0.f ? v : expf(v) - 1.f;
        v += res[(size_t)row * D + f];
        vals[i] = v;
        s += v; s2 += v * v;
    }
#pragma unroll
    for (int w = 16; w >= 1; w >>= 1) {
        s  += __shfl_xor(s, w, 32);
        s2 += __shfl_xor(s2, w, 32);
    }
    __shared__ float sh1[8], sh2[8];
    int wid = tid >> 5;
    if ((tid & 31) == 0) { sh1[wid] = s; sh2[wid] = s2; }
    __syncthreads();
    if (tid < 32) {
        float a  = (tid < 8) ? sh1[tid] : 0.f;
        float b2 = (tid < 8) ? sh2[tid] : 0.f;
#pragma unroll
        for (int w = 4; w >= 1; w >>= 1) {
            a += __shfl_xor(a, w, 32);
            b2 += __shfl_xor(b2, w, 32);
        }
        if (tid == 0) { sh1[0] = a; sh2[0] = b2; }
    }
    __syncthreads();
    float mean = sh1[0] / (float)D;
    float var  = sh2[0] / (float)D - mean * mean;
    float inv  = rsqrtf(var + LNEPS);
    for (int i = 0; i < per; ++i) {
        int f = tid + (i << 8);
        out[(size_t)row * D + f] = (vals[i] - mean) * inv * g[f] + be[f];
    }
}

// ---------------------------------------------------------------------------
// Host orchestration
// ---------------------------------------------------------------------------
static inline int host_fenc(float f) {
    unsigned u; memcpy(&u, &f, 4);
    int b = (int)u;
    return b >= 0 ? b : (int)(0x80000000u - u);
}
static inline int cdiv(long long a, long long b) { return (int)((a + b - 1) / b); }

extern "C" void kernel_launch(void* const* d_in, const int* in_sizes, int n_in,
                              void* d_out, int out_size, void* d_ws, size_t ws_size,
                              hipStream_t stream) {
    (void)in_sizes; (void)n_in; (void)out_size; (void)ws_size;
    const int N = NNODES, E0 = NEDGE, Ea = EALL;

    // ---- inputs (setup_inputs order)
    const float* x     = (const float*)d_in[0];
    const int*   ei    = (const int*)d_in[1];
    const float* W_in  = (const float*)d_in[2];
    const float* b_in  = (const float*)d_in[3];
    const float* Wl0   = (const float*)d_in[4];
    const float* bl0   = (const float*)d_in[5];
    const float* Wr0   = (const float*)d_in[6];
    const float* br0   = (const float*)d_in[7];
    const float* att0  = (const float*)d_in[8];
    const float* bc0   = (const float*)d_in[9];
    const float* g0    = (const float*)d_in[10];
    const float* be0   = (const float*)d_in[11];
    const float* Wl1   = (const float*)d_in[12];
    const float* bl1   = (const float*)d_in[13];
    const float* Wr1   = (const float*)d_in[14];
    const float* br1   = (const float*)d_in[15];
    const float* att1  = (const float*)d_in[16];
    const float* bc1   = (const float*)d_in[17];
    const float* g1    = (const float*)d_in[18];
    const float* be1   = (const float*)d_in[19];
    const float* Wl2   = (const float*)d_in[20];
    const float* bl2   = (const float*)d_in[21];
    const float* Wr2   = (const float*)d_in[22];
    const float* br2   = (const float*)d_in[23];
    const float* att2  = (const float*)d_in[24];
    const float* bc2   = (const float*)d_in[25];
    const float* g2    = (const float*)d_in[26];
    const float* be2   = (const float*)d_in[27];
    const float* Wres2 = (const float*)d_in[28];
    const float* bres2 = (const float*)d_in[29];

    const int* srcE = ei;
    const int* dstE = ei + E0;

    // ---- workspace carve
    size_t off = 0;
    char* base = (char*)d_ws;
    auto carve = [&](size_t bytes) -> void* {
        void* r = base + off;
        off += (bytes + 255) & ~(size_t)255;
        return r;
    };
    float*    h     = (float*)   carve((size_t)N * HDIM * 4);
    float*    xl    = (float*)   carve((size_t)N * HDIM * 4);
    float*    xr    = (float*)   carve((size_t)N * HDIM * 4);   // reused as aggr
    float*    resb  = (float*)   carve((size_t)N * OUTD * 4);
    float*    ebuf  = (float*)   carve((size_t)Ea * HHEADS * 4);
    int*      emax  = (int*)     carve((size_t)N * HHEADS * 4);
    float*    den   = (float*)   carve((size_t)N * HHEADS * 4);
    ushort_t* hb    = (ushort_t*)carve((size_t)N * HDIM * 2);
    ushort_t* WtIn  = (ushort_t*)carve((size_t)HDIM * DIN * 2);
    ushort_t* WtL0  = (ushort_t*)carve((size_t)HDIM * HDIM * 2);
    ushort_t* WtR0  = (ushort_t*)carve((size_t)HDIM * HDIM * 2);
    ushort_t* WtL1  = (ushort_t*)carve((size_t)HDIM * HDIM * 2);
    ushort_t* WtR1  = (ushort_t*)carve((size_t)HDIM * HDIM * 2);
    ushort_t* WtL2  = (ushort_t*)carve((size_t)OUTD * HDIM * 2);
    ushort_t* WtR2  = (ushort_t*)carve((size_t)OUTD * HDIM * 2);
    ushort_t* WtRs2 = (ushort_t*)carve((size_t)OUTD * HDIM * 2);

    const int ENC_NEG_INF = host_fenc(-INFINITY);

    auto wconv = [&](const float* W, ushort_t* Wt, int K, int Nn) {
        int n = K * Nn;
        wconv_kernel<<<cdiv(n, 256), 256, 0, stream>>>(W, Wt, K, Nn);
    };
    auto gemm = [&](const ushort_t* Ab, const ushort_t* Wt, const float* bias,
                    float* Cout, int M, int Nn, int K, int act) {
        dim3 grid(cdiv(M, BM), Nn / BN);
        gemm_bf16_kernel<<<grid, 256, 0, stream>>>(Ab, Wt, bias, Cout, M, Nn, K, act);
    };

    // ---- convert weights (bf16, transposed) — every call, deterministic
    wconv(W_in, WtIn, DIN, HDIM);
    wconv(Wl0, WtL0, HDIM, HDIM);  wconv(Wr0, WtR0, HDIM, HDIM);
    wconv(Wl1, WtL1, HDIM, HDIM);  wconv(Wr1, WtR1, HDIM, HDIM);
    wconv(Wl2, WtL2, HDIM, OUTD);  wconv(Wr2, WtR2, HDIM, OUTD);
    wconv(Wres2, WtRs2, HDIM, OUTD);

    // ---- input projection: h = elu(x @ W_in + b_in)
    conv_bf16_kernel<<<cdiv((long long)N * DIN, 256), 256, 0, stream>>>(x, hb, N * DIN);
    gemm(hb, WtIn, b_in, h, N, HDIM, DIN, 1);

    // ---- attention layers 0,1 (identity residual)
    const float* WLt[2][2] = {{nullptr, nullptr}, {nullptr, nullptr}}; (void)WLt;
    for (int layer = 0; layer < 2; ++layer) {
        const ushort_t* WtL = layer ? WtL1 : WtL0;
        const ushort_t* WtR = layer ? WtR1 : WtR0;
        const float* bl  = layer ? bl1  : bl0;
        const float* br  = layer ? br1  : br0;
        const float* att = layer ? att1 : att0;
        const float* bc  = layer ? bc1  : bc0;
        const float* gg  = layer ? g1   : g0;
        const float* bb  = layer ? be1  : be0;

        conv_bf16_kernel<<<cdiv((long long)N * HDIM, 256), 256, 0, stream>>>(h, hb, N * HDIM);
        gemm(hb, WtL, bl, xl, N, HDIM, HDIM, 0);
        gemm(hb, WtR, br, xr, N, HDIM, HDIM, 0);

        fill_i32_kernel<<<cdiv(N * HHEADS, 256), 256, 0, stream>>>(emax, ENC_NEG_INF, N * HHEADS);
        fill_f32_kernel<<<cdiv(N * HHEADS, 256), 256, 0, stream>>>(den, 0.f, N * HHEADS);

        edge_logits_kernel<HDIM, CDIM><<<cdiv((long long)Ea * 32, 256), 256, 0, stream>>>(
            xl, xr, att, srcE, dstE, E0, Ea, ebuf, emax);
        edge_exp_kernel<<<cdiv((long long)Ea * HHEADS, 256), 256, 0, stream>>>(
            ebuf, emax, den, dstE, E0, Ea, HHEADS);

        // xr dead from here -> reuse as aggregation buffer
        fill_f32_kernel<<<cdiv((long long)N * HDIM, 256), 256, 0, stream>>>(xr, 0.f, N * HDIM);
        edge_scatter_kernel<<<cdiv((long long)Ea * (HDIM / 4), 256), 256, 0, stream>>>(
            xl, ebuf, den, srcE, dstE, E0, Ea, HDIM, CDIM, xr);

        elu_res_ln_kernel<<<N, 256, 0, stream>>>(xr, bc, h, gg, bb, h, HDIM);
    }

    // ---- layer 2: 1024 -> 256, single head, linear residual
    conv_bf16_kernel<<<cdiv((long long)N * HDIM, 256), 256, 0, stream>>>(h, hb, N * HDIM);
    gemm(hb, WtL2,  bl2,   xl,   N, OUTD, HDIM, 0);
    gemm(hb, WtR2,  br2,   xr,   N, OUTD, HDIM, 0);
    gemm(hb, WtRs2, bres2, resb, N, OUTD, HDIM, 0);

    fill_i32_kernel<<<cdiv(N, 256), 256, 0, stream>>>(emax, ENC_NEG_INF, N);
    fill_f32_kernel<<<cdiv(N, 256), 256, 0, stream>>>(den, 0.f, N);

    edge_logits_kernel<OUTD, OUTD><<<cdiv((long long)Ea * 32, 256), 256, 0, stream>>>(
        xl, xr, att2, srcE, dstE, E0, Ea, ebuf, emax);
    edge_exp_kernel<<<cdiv(Ea, 256), 256, 0, stream>>>(
        ebuf, emax, den, dstE, E0, Ea, 1);

    fill_f32_kernel<<<cdiv((long long)N * OUTD, 256), 256, 0, stream>>>(xr, 0.f, N * OUTD);
    edge_scatter_kernel<<<cdiv((long long)Ea * (OUTD / 4), 256), 256, 0, stream>>>(
        xl, ebuf, den, srcE, dstE, E0, Ea, OUTD, OUTD, xr);

    elu_res_ln_kernel<<<N, 256, 0, stream>>>(xr, bc2, resb, g2, be2, (float*)d_out, OUTD);
}
